// TkVectorQuantizerEMA_46042049413922
// MI455X (gfx1250) — matrix operand order repaired and checked by reading, not compiled, not run on hardware
//
#include <hip/hip_runtime.h>

typedef __attribute__((ext_vector_type(16))) _Float16 v16h;
typedef __attribute__((ext_vector_type(8)))  float    v8f;

#define NUM_K   4096
#define DIM     256
#define T_LEN   1024
#define B_SZ    64
#define N_ROWS  (B_SZ * T_LEN)          // 65536 rows
#define KTILES  (NUM_K / 16)            // 256

#if __has_builtin(__builtin_amdgcn_tensor_load_to_lds) && \
    __has_builtin(__builtin_amdgcn_s_wait_tensorcnt)
#define USE_TDM 1
#else
#define USE_TDM 0
#endif

#if USE_TDM
// ---------------------------------------------------------------------------
// Issue one TDM DMA: copy an 8KB codebook tile (contiguous) into LDS.
// D# per CDNA5 ISA ch.8: group0 {count=1, lds_addr, global_addr, type=2},
// group1 {data_size=8B, tensor_dim0 = huge (no OOB clamp), tile_dim0 = 1024
// units of 8B, tile_dim1 = 1}, groups 2/3 zero (<=2D tensor mode).
// ---------------------------------------------------------------------------
__device__ __forceinline__ void tdm_load_tile_8k(const void* gsrc, unsigned ldsOff) {
  typedef __attribute__((ext_vector_type(4))) unsigned v4u;
  typedef __attribute__((ext_vector_type(8))) int      v8i;
  typedef __attribute__((ext_vector_type(4))) int      v4i;
  const unsigned long long ga = (unsigned long long)(size_t)gsrc;
  const v4u g0 = { 1u,                                  // count=1, is_restore=0
                   ldsOff,                              // lds_addr (bytes)
                   (unsigned)ga,                        // global_addr[31:0]
                   (unsigned)(ga >> 32) | 0x80000000u };// addr[56:32] | type=2
  const v8i g1 = { (int)(3u << 16),                     // data_size = 8 bytes
                   0,                                   // tensor_dim0 lo16<<16
                   0x00014000,                          // dim0 hi=0x4000, dim1=1
                   0x04000000,                          // tile_dim0 = 1024 <<16
                   1,                                   // tile_dim1 = 1
                   1024, 0, 0 };                        // dim0_stride = 1024
  const v4i gz = { 0, 0, 0, 0 };
#if __clang_major__ >= 23
  const v8i gz8 = { 0, 0, 0, 0, 0, 0, 0, 0 };
  __builtin_amdgcn_tensor_load_to_lds(g0, g1, gz, gz, gz8, 0);
#else
  __builtin_amdgcn_tensor_load_to_lds(g0, g1, gz, gz, 0);
#endif
}
#endif

// ---------------------------------------------------------------------------
// Kernel 1: L2-normalize codebook rows (fp32) into an f16 copy for WMMA B.
// ---------------------------------------------------------------------------
__global__ __launch_bounds__(256) void k_wnorm(const float* __restrict__ weight,
                                               _Float16* __restrict__ wnorm) {
  const int k = blockIdx.x;
  const int d = threadIdx.x;
  float w  = weight[k * DIM + d];
  float ss = w * w;
  #pragma unroll
  for (int s = 1; s < 32; s <<= 1) ss += __shfl_xor(ss, s, 32);
  __shared__ float red[8];
  if ((threadIdx.x & 31) == 0) red[threadIdx.x >> 5] = ss;
  __syncthreads();
  float tot = 0.f;
  #pragma unroll
  for (int i = 0; i < 8; ++i) tot += red[i];
  const float recip = 1.0f / fmaxf(sqrtf(tot), 1e-6f);
  wnorm[k * DIM + d] = (_Float16)(w * recip);
}

// ---------------------------------------------------------------------------
// Kernel 2: fused cos-sim GEMM (WMMA f16->f32) + argmax + gather + EMA stats.
// One wave per 16-row z tile; 8 waves/block share B tiles via TDM-staged LDS.
// ---------------------------------------------------------------------------
__global__ __launch_bounds__(256) void k_vq_main(
    const float* __restrict__ z, const float* __restrict__ weight,
    const _Float16* __restrict__ wnorm,
    float* __restrict__ out_q, float* __restrict__ out_codes,
    float* __restrict__ dw, float* __restrict__ counts,
    float* __restrict__ loss_sum)
{
  __shared__ alignas(32) _Float16 sB[2][16 * DIM];  // 2 x 8KB double-buffered B tile
  __shared__ int sIdx[8][16];

  const int tid  = threadIdx.x;
  const int wave = tid >> 5;
  const int lane = tid & 31;
  const int m    = lane & 15;   // row-in-tile (A/C) and column-in-tile (B/C)
  const int half = lane >> 4;

  const int rowBase = blockIdx.x * 128 + wave * 16;
  const int n = rowBase + m;
  const int b = n >> 10;            // n / T
  const int t = n & (T_LEN - 1);    // n % T
  const float* zrow = z + (size_t)b * (DIM * T_LEN) + t;   // element d at +d*T

  // ---- Build A fragments (raw f16) and row sum-of-squares ----------------
  // A 16x32, this code's contiguous k order (same as B): lane(half) holds K = 32c + 16*half + {0..15}
  v16h a[8];
  float ss = 0.f;
  #pragma unroll
  for (int c = 0; c < 8; ++c) {
    const int d0 = c * 32 + half * 16;
    #pragma unroll
    for (int e = 0; e < 8; ++e) {
      float x = zrow[(d0 + e) * T_LEN];
      ss += x * x;
      a[c][e] = (_Float16)x;
      float y = zrow[(d0 + 8 + e) * T_LEN];
      ss += y * y;
      a[c][e + 8] = (_Float16)y;
    }
  }
  ss += __shfl_xor(ss, 16, 32);     // combine the two half-lanes of this row
  const float recip = 1.0f / fmaxf(sqrtf(ss), 1e-6f);
  {
    const _Float16 hr = (_Float16)recip;
    const v16h hv = {hr,hr,hr,hr,hr,hr,hr,hr,hr,hr,hr,hr,hr,hr,hr,hr};
    #pragma unroll
    for (int c = 0; c < 8; ++c) a[c] *= hv;   // v_pk_mul_f16 scale -> normalized A
  }

  float bestv[8];
  int   besti[8];
  #pragma unroll
  for (int r = 0; r < 8; ++r) { bestv[r] = -__builtin_inff(); besti[r] = 0; }

#if USE_TDM
  const unsigned ldsBase = (unsigned)(size_t)&sB[0][0];  // low 32 bits = LDS offset
  if (wave == 0) {                                       // preload tile 0 via TDM
    tdm_load_tile_8k(wnorm, ldsBase);
    __builtin_amdgcn_s_wait_tensorcnt(0);
  }
  __syncthreads();
#else
  {                                 // preload tile 0 (512 uint4 = 8KB)
    const uint4* src = (const uint4*)wnorm;
    uint4* dst = (uint4*)&sB[0][0];
    dst[tid]       = src[tid];
    dst[tid + 256] = src[tid + 256];
  }
  __syncthreads();
#endif

  for (int kt = 0; kt < KTILES; ++kt) {
    const int cur = kt & 1;
#if USE_TDM
    if ((kt + 1 < KTILES) && (wave == 0))   // TDM-stage next tile into other buf
      tdm_load_tile_8k(wnorm + (size_t)(kt + 1) * 16 * DIM,
                       ldsBase + (unsigned)((cur ^ 1) * 16 * DIM * 2));
#else
    if (kt + 1 < KTILES) {
      const uint4* src = (const uint4*)(wnorm + (size_t)(kt + 1) * 16 * DIM);
      uint4* dst = (uint4*)&sB[cur ^ 1][0];
      dst[tid]       = src[tid];
      dst[tid + 256] = src[tid + 256];
    }
#endif
    if (kt + 2 < KTILES)                    // global_prefetch_b8 into GL2
      __builtin_prefetch(wnorm + (size_t)(kt + 2) * 16 * DIM + tid * 16, 0, 0);

    // ---- 16x16 tile of cos-sim: 8 chained WMMA over D=256 ----------------
    v8f acc = {};
    #pragma unroll
    for (int c = 0; c < 8; ++c) {
      // B 32x16, this code's contiguous k order (same as A): lane(half) holds K = 32c + 16*half + {0..15} of column m
      const v16h bf = *(const v16h*)&sB[cur][m * DIM + c * 32 + half * 16];
      acc = __builtin_amdgcn_wmma_f32_16x16x32_f16(false, a[c], false, bf,
                                                   (short)0, acc, false, false);
    }

    // ---- running argmax: this lane holds column codeN, rows M = r + 8*half
    const int  codeN = kt * 16 + m;
    const bool excl  = (kt == 0) & (m == 0);  // reserved code 0 -> -inf
    #pragma unroll
    for (int r = 0; r < 8; ++r) {
      float v = excl ? -__builtin_inff() : acc[r];
      if (v > bestv[r]) { bestv[r] = v; besti[r] = codeN; }
    }
#if USE_TDM
    if (wave == 0) __builtin_amdgcn_s_wait_tensorcnt(0);
#endif
    __syncthreads();
  }

  // ---- cross-lane argmax reduction within each 16-lane half ---------------
  #pragma unroll
  for (int r = 0; r < 8; ++r) {
    float v = bestv[r]; int i = besti[r];
    #pragma unroll
    for (int s = 1; s < 16; s <<= 1) {
      float ov = __shfl_xor(v, s, 32);
      int   oi = __shfl_xor(i, s, 32);
      if (ov > v || (ov == v && oi < i)) { v = ov; i = oi; }   // first-index ties
    }
    bestv[r] = v; besti[r] = i;
  }

  if (lane == 0) {                 // rows 0..7 of this wave's tile
    #pragma unroll
    for (int r = 0; r < 8; ++r) {
      sIdx[wave][r] = besti[r];
      out_codes[rowBase + r] = (float)besti[r];
    }
  } else if (lane == 16) {         // rows 8..15
    #pragma unroll
    for (int r = 0; r < 8; ++r) {
      sIdx[wave][8 + r] = besti[r];
      out_codes[rowBase + 8 + r] = (float)besti[r];
    }
  }
  __syncthreads();

  // ---- epilogue: gather q, write straight-through output, stats -----------
  const int myIdx = sIdx[wave][m];
  const float* wrow = weight + (size_t)myIdx * DIM;
  float* qrow = out_q + (size_t)b * (DIM * T_LEN) + t;
  float lossp = 0.f;
  const int dBase = half * 128;    // the two half-lanes split this row's D range
  for (int j = 0; j < 128; ++j) {
    const int d  = dBase + j;
    const float zz = zrow[d * T_LEN];
    const float q  = wrow[d];
    const float df = q - zz;
    lossp += df * df;
    qrow[d * T_LEN] = q;                                   // quantized [B,D,T]
    atomicAdd(&dw[(size_t)myIdx * DIM + d], zz * recip);   // segment_sum(z_norm)
  }
  #pragma unroll
  for (int s = 1; s < 32; s <<= 1) lossp += __shfl_xor(lossp, s, 32);
  if (lane == 0) atomicAdd(loss_sum, lossp);
  if (lane < 16) atomicAdd(&counts[myIdx], 1.0f);          // one count per row
}

// ---------------------------------------------------------------------------
// Kernel 3: new_cs = decay*ema_cs + (1-decay)*counts; n = sum(new_cs); loss.
// ---------------------------------------------------------------------------
__global__ __launch_bounds__(256) void k_f1(const float* __restrict__ ema_cs,
                                            const float* __restrict__ counts,
                                            const float* __restrict__ loss_sum,
                                            float* __restrict__ out_cs,
                                            float* __restrict__ out_loss,
                                            float* __restrict__ n_sum) {
  const int k = blockIdx.x * blockDim.x + threadIdx.x;
  if (k < NUM_K) {
    float v = 0.99f * ema_cs[k] + 0.01f * counts[k];
    if (k == 0) v = 0.f;
    out_cs[k] = v;
    atomicAdd(n_sum, v);
  }
  if (k == 0)
    out_loss[0] = 0.25f * loss_sum[0] / (float)(N_ROWS * DIM);
}

// ---------------------------------------------------------------------------
// Kernel 4: new_ema_w, cluster-size rescale, L2-normalized new_weight.
// ---------------------------------------------------------------------------
__global__ __launch_bounds__(256) void k_f2(const float* __restrict__ ema_w,
                                            const float* __restrict__ dw,
                                            const float* __restrict__ new_cs,
                                            const float* __restrict__ n_sum,
                                            float* __restrict__ out_w,
                                            float* __restrict__ out_ema_w) {
  const int k = blockIdx.x;
  const int d = threadIdx.x;
  float ew = 0.99f * ema_w[k * DIM + d] + 0.01f * dw[k * DIM + d];
  if (k == 0) ew = 0.f;
  out_ema_w[k * DIM + d] = ew;

  const float nn = n_sum[0];
  const float cs = (new_cs[k] + 1e-5f) / (nn + (float)NUM_K * 1e-5f) * nn;
  const float tv = ew / cs;

  float ss = tv * tv;
  #pragma unroll
  for (int s = 1; s < 32; s <<= 1) ss += __shfl_xor(ss, s, 32);
  __shared__ float red[8];
  if ((threadIdx.x & 31) == 0) red[threadIdx.x >> 5] = ss;
  __syncthreads();
  float tot = 0.f;
  #pragma unroll
  for (int i = 0; i < 8; ++i) tot += red[i];
  float w = tv / fmaxf(sqrtf(tot), 1e-6f);
  if (k == 0) w = 0.f;
  out_w[k * DIM + d] = w;
}

// ---------------------------------------------------------------------------
extern "C" void kernel_launch(void* const* d_in, const int* in_sizes, int n_in,
                              void* d_out, int out_size, void* d_ws, size_t ws_size,
                              hipStream_t stream) {
  (void)in_sizes; (void)n_in; (void)out_size; (void)ws_size;
  const float* z      = (const float*)d_in[0];   // [64, 256, 1024]
  const float* weight = (const float*)d_in[1];   // [4096, 256]
  const float* ema_cs = (const float*)d_in[2];   // [4096]
  const float* ema_w  = (const float*)d_in[3];   // [4096, 256]

  float* out = (float*)d_out;                    // tuple concatenated flat
  float* out_q     = out;                        // 16777216  quantized [B,D,T]
  float* out_loss  = out + 16777216;             // 1         loss
  float* out_codes = out + 16777217;             // 65536     codes (as float)
  float* out_w     = out + 16842753;             // 1048576   new_weight
  float* out_cs    = out + 17891329;             // 4096      new_cs
  float* out_emw   = out + 17895425;             // 1048576   new_ema_w

  char* ws = (char*)d_ws;
  _Float16* wnorm  = (_Float16*)ws;                        // 2 MB  f16 codebook
  float* dw        = (float*)(ws + (2u << 20));            // 4 MB  segment sums
  float* counts    = (float*)(ws + (6u << 20));            // 16 KB
  float* loss_sum  = (float*)(ws + (6u << 20) + 16384);    // 4 B
  float* n_sum     = loss_sum + 1;                         // 4 B

  // zero the atomic accumulators every call (ws is not re-poisoned/zeroed)
  hipMemsetAsync(dw, 0, (4u << 20) + 16384 + 8, stream);

  k_wnorm  <<<dim3(NUM_K),        dim3(256), 0, stream>>>(weight, wnorm);
  k_vq_main<<<dim3(N_ROWS / 128), dim3(256), 0, stream>>>(z, weight, wnorm,
                                                          out_q, out_codes,
                                                          dw, counts, loss_sum);
  k_f1     <<<dim3(NUM_K / 256),  dim3(256), 0, stream>>>(ema_cs, counts, loss_sum,
                                                          out_cs, out_loss, n_sum);
  k_f2     <<<dim3(NUM_K),        dim3(256), 0, stream>>>(ema_w, dw, out_cs, n_sum,
                                                          out_w, out_emw);
}
